// Attention_63900523430102
// MI455X (gfx1250) — matrix operand order repaired and checked by reading, not compiled, not run on hardware
//
#include <hip/hip_runtime.h>
#include <hip/hip_bf16.h>

typedef __bf16 v16bf __attribute__((ext_vector_type(16)));
typedef float  v8f   __attribute__((ext_vector_type(8)));
typedef int    v4i   __attribute__((ext_vector_type(4)));

#define D_MODEL  1024
#define NHEADS   16
#define HEAD_DIM 64
#define SEQLEN   2048
#define BS       4
#define MROWS    (SEQLEN * BS)   /* 8192 */
#define NCHUNK   (SEQLEN / 64)   /* 32 key chunks */

// ---- CDNA5 async global->LDS path (guarded; falls back to sync copies) ----
#if defined(__HIP_DEVICE_COMPILE__) &&                                  \
    __has_builtin(__builtin_amdgcn_global_load_async_to_lds_b128) &&    \
    __has_builtin(__builtin_amdgcn_s_wait_asynccnt)
#define HAVE_ASYNC_LDS 1
#else
#define HAVE_ASYNC_LDS 0
#endif

#if HAVE_ASYNC_LDS
// builtin signature (from hipcc diagnostic): (v4i as1*, v4i as3*, imm, imm)
#define ASYNC_B128(gp, lp)                                              \
  __builtin_amdgcn_global_load_async_to_lds_b128(                       \
      (__attribute__((address_space(1))) v4i*)(gp),                     \
      (__attribute__((address_space(3))) v4i*)(lp), 0, 0)
// copy 64 bytes (32 bf16) global -> LDS, 4x b128 per lane
static __device__ __forceinline__ void async_cp32e(const unsigned short* g,
                                                   unsigned short* l) {
  ASYNC_B128(g,      l);
  ASYNC_B128(g + 8,  l + 8);
  ASYNC_B128(g + 16, l + 16);
  ASYNC_B128(g + 24, l + 24);
}
#endif

static __device__ __forceinline__ unsigned short f2bf(float f) {
  unsigned u = __builtin_bit_cast(unsigned, f);
  u += 0x7FFFu + ((u >> 16) & 1u);               // round-to-nearest-even
  return (unsigned short)(u >> 16);
}
static __device__ __forceinline__ float bf2f(unsigned short h) {
  unsigned u = ((unsigned)h) << 16;
  return __builtin_bit_cast(float, u);
}
static __device__ __forceinline__ unsigned pk2(float a, float b) {
  return (unsigned)f2bf(a) | ((unsigned)f2bf(b) << 16);
}

// A fragment (16x32 bf16): lane holds K = (lane>>4)*16 .. +15 contiguous.
static __device__ __forceinline__ v16bf ld_fragA(const unsigned short* p) {
  uint4 lo = *(const uint4*)(p);
  uint4 hi = *(const uint4*)(p + 8);
  v16bf r;
  *((uint4*)&r)     = lo;
  *((uint4*)&r + 1) = hi;
  return r;
}
// B fragment (32x16 bf16): lane = column, K = (lane>>4)*16 .. +15 contiguous.
static __device__ __forceinline__ v16bf ld_fragB(const unsigned short* p) {
  uint4 lo = *(const uint4*)(p);
  uint4 hi = *(const uint4*)(p + 8);
  v16bf r;
  *((uint4*)&r)     = lo;
  *((uint4*)&r + 1) = hi;
  return r;
}

// ---------------------------------------------------------------------------
// NT GEMM: Out[M,N] = A[M,K] * W[N,K]^T, bf16 WMMA, fp32 accumulate.
// BM=64 BN=128 BK=32, 256 threads = 8 waves (2x4), 32x32 per wave.
// 2-stage pipeline: next tile prefetched into registers during WMMA.
// ---------------------------------------------------------------------------
template<bool A_BF16, bool OUT_F32>
__global__ __launch_bounds__(256) void gemm_nt_wmma(const void* __restrict__ Ap,
                                                    const float* __restrict__ W,
                                                    void* __restrict__ Outp,
                                                    int M, int N, int K) {
  __shared__ __attribute__((aligned(16))) unsigned short lA[64 * 32];
  __shared__ __attribute__((aligned(16))) unsigned short lB[128 * 32];

  const int tid  = threadIdx.x;
  const int lane = tid & 31;
  const int ln   = lane & 15;
  const int hl   = lane >> 4;
  const int wid  = tid >> 5;

  const int mb = blockIdx.y * 64;
  const int nb = blockIdx.x * 128;
  const int wm = (wid >> 2) * 32;
  const int wn = (wid & 3) * 32;

  const int arow = tid >> 2, acol = (tid & 3) * 8;   // A: 64x32, 8 elem/thread
  const int brow = tid >> 1, bcol = (tid & 1) * 16;  // B: 128x32, 16 elem/thread

  // staging registers (tile t+1 loaded while tile t computes)
  uint4  sAb;        // A when bf16
  float4 sAf0, sAf1; // A when fp32
  float4 sB0, sB1, sB2, sB3;

  if (A_BF16) {
    sAb = *(const uint4*)((const unsigned short*)Ap + (size_t)(mb + arow) * K + acol);
  } else {
    const float4* s = (const float4*)((const float*)Ap + (size_t)(mb + arow) * K + acol);
    sAf0 = s[0]; sAf1 = s[1];
  }
  {
    const float4* s = (const float4*)(W + (size_t)(nb + brow) * K + bcol);
    sB0 = s[0]; sB1 = s[1]; sB2 = s[2]; sB3 = s[3];
  }

  const v8f zf = {0.f, 0.f, 0.f, 0.f, 0.f, 0.f, 0.f, 0.f};
  v8f acc[2][2] = {{zf, zf}, {zf, zf}};

  for (int kt = 0; kt < K; kt += 32) {
    __syncthreads();                 // previous compute done reading LDS
    // commit staged tile (convert fp32 -> bf16 here)
    if (A_BF16) {
      *(uint4*)&lA[arow * 32 + acol] = sAb;
    } else {
      uint4 p;
      p.x = pk2(sAf0.x, sAf0.y); p.y = pk2(sAf0.z, sAf0.w);
      p.z = pk2(sAf1.x, sAf1.y); p.w = pk2(sAf1.z, sAf1.w);
      *(uint4*)&lA[arow * 32 + acol] = p;
    }
    {
      uint4 p0, p1;
      p0.x = pk2(sB0.x, sB0.y); p0.y = pk2(sB0.z, sB0.w);
      p0.z = pk2(sB1.x, sB1.y); p0.w = pk2(sB1.z, sB1.w);
      p1.x = pk2(sB2.x, sB2.y); p1.y = pk2(sB2.z, sB2.w);
      p1.z = pk2(sB3.x, sB3.y); p1.w = pk2(sB3.z, sB3.w);
      *(uint4*)&lB[brow * 32 + bcol]     = p0;
      *(uint4*)&lB[brow * 32 + bcol + 8] = p1;
    }
    __syncthreads();

    // prefetch next tile into registers (overlaps with WMMA below)
    const int kn = kt + 32;
    if (kn < K) {
      if (A_BF16) {
        sAb = *(const uint4*)((const unsigned short*)Ap + (size_t)(mb + arow) * K + kn + acol);
      } else {
        const float4* s = (const float4*)((const float*)Ap + (size_t)(mb + arow) * K + kn + acol);
        sAf0 = s[0]; sAf1 = s[1];
      }
      const float4* s = (const float4*)(W + (size_t)(nb + brow) * K + kn + bcol);
      sB0 = s[0]; sB1 = s[1]; sB2 = s[2]; sB3 = s[3];
    }

    v16bf af[2], bfr[2];
#pragma unroll
    for (int tm = 0; tm < 2; ++tm)
      af[tm] = ld_fragA(&lA[(wm + tm * 16 + ln) * 32 + hl * 16]);
#pragma unroll
    for (int tn = 0; tn < 2; ++tn)
      bfr[tn] = ld_fragB(&lB[(wn + tn * 16 + ln) * 32 + hl * 16]);
#pragma unroll
    for (int tm = 0; tm < 2; ++tm)
#pragma unroll
      for (int tn = 0; tn < 2; ++tn)
        acc[tm][tn] = __builtin_amdgcn_wmma_f32_16x16x32_bf16(
            false, af[tm], false, bfr[tn], (short)0, acc[tm][tn], false, false);
  }

#pragma unroll
  for (int tm = 0; tm < 2; ++tm)
#pragma unroll
    for (int tn = 0; tn < 2; ++tn)
#pragma unroll
      for (int j = 0; j < 8; ++j) {
        int row = mb + wm + tm * 16 + j + hl * 8;
        int col = nb + wn + tn * 16 + ln;
        float val = acc[tm][tn][j];
        if (OUT_F32)
          ((float*)Outp)[(size_t)row * N + col] = val;
        else
          ((unsigned short*)Outp)[(size_t)row * N + col] = f2bf(val);
      }
}

// ---------------------------------------------------------------------------
// RoPE applied in-place on bf16 xq/xk.
// ---------------------------------------------------------------------------
__global__ __launch_bounds__(256) void rope_kernel(unsigned short* __restrict__ xq,
                                                   unsigned short* __restrict__ xk,
                                                   const float* __restrict__ fc) {
  int idx = blockIdx.x * 256 + threadIdx.x;
  const int total = MROWS * (D_MODEL / 2);
  if (idx >= total) return;
  int r = idx >> 9;          // row = s*BS + b (512 pairs per row)
  int p = idx & 511;
  int h = p >> 5;
  int i = p & 31;
  int s = r >> 2;            // / BS
  float c  = fc[s * 64 + i * 2];
  float sn = fc[s * 64 + i * 2 + 1];
  size_t base = (size_t)r * D_MODEL + h * 64 + 2 * i;
  {
    float x0 = bf2f(xq[base]), x1 = bf2f(xq[base + 1]);
    xq[base]     = f2bf(x0 * c - x1 * sn);
    xq[base + 1] = f2bf(x1 * c + x0 * sn);
  }
  {
    float x0 = bf2f(xk[base]), x1 = bf2f(xk[base + 1]);
    xk[base]     = f2bf(x0 * c - x1 * sn);
    xk[base + 1] = f2bf(x1 * c + x0 * sn);
  }
}

// ---------------------------------------------------------------------------
// One-shot V transpose: xv (l-major per head) -> vt[b,h][d][l], so the
// attention inner loop streams both K and V^T with pure contiguous copies.
// ---------------------------------------------------------------------------
__global__ __launch_bounds__(128) void transpose_v_kernel(const unsigned short* __restrict__ xv,
                                                          unsigned short* __restrict__ vt) {
  __shared__ __attribute__((aligned(16))) unsigned short lT[64 * 64];
  const int lc = blockIdx.x;                 // key chunk
  const int b  = blockIdx.y >> 4;
  const int h  = blockIdx.y & 15;
  const int tid = threadIdx.x;
  {
    int li = tid >> 1, d0 = (tid & 1) * 32;
    size_t g = ((size_t)(lc * 64 + li) * BS + b) * D_MODEL + h * 64 + d0;
    const uint4* vs = (const uint4*)(xv + g);
    uint4 vv[4];
#pragma unroll
    for (int i = 0; i < 4; ++i) vv[i] = vs[i];
    const unsigned short* vu = (const unsigned short*)vv;
#pragma unroll
    for (int e = 0; e < 32; ++e)
      lT[(d0 + e) * 64 + li] = vu[e];
  }
  __syncthreads();
  {
    int d = tid >> 1, l0 = (tid & 1) * 32;
    size_t o = ((size_t)blockIdx.y * 64 + d) * SEQLEN + lc * 64 + l0;
    uint4* dst = (uint4*)(vt + o);
    const uint4* src = (const uint4*)&lT[d * 64 + l0];
#pragma unroll
    for (int i = 0; i < 4; ++i) dst[i] = src[i];
  }
}

// ---------------------------------------------------------------------------
// Per-chunk flash attention compute (64 keys): S = Q*K^T, online softmax,
// O += P*V.  Wave w owns query rows 16w..16w+15.
// ---------------------------------------------------------------------------
static __device__ __forceinline__ void attn_chunk(const unsigned short* __restrict__ lK,
                                                  const unsigned short* __restrict__ lVt,
                                                  unsigned short* __restrict__ lP,
                                                  const v16bf* qa, v8f* O,
                                                  float* rowm, float* rowl,
                                                  int w, int ln, int hl) {
  const v8f zf = {0.f, 0.f, 0.f, 0.f, 0.f, 0.f, 0.f, 0.f};
  v8f sa[4] = {zf, zf, zf, zf};
#pragma unroll
  for (int kt = 0; kt < 2; ++kt)
#pragma unroll
    for (int nt = 0; nt < 4; ++nt) {
      v16bf kb = ld_fragB(&lK[(nt * 16 + ln) * 64 + kt * 32 + hl * 16]);
      sa[nt] = __builtin_amdgcn_wmma_f32_16x16x32_bf16(
          false, qa[kt], false, kb, (short)0, sa[nt], false, false);
    }
#pragma unroll
  for (int nt = 0; nt < 4; ++nt)
#pragma unroll
    for (int j = 0; j < 8; ++j) sa[nt][j] *= 0.125f;   // 1/sqrt(64)

  float cm[8];
#pragma unroll
  for (int j = 0; j < 8; ++j) {
    cm[j] = fmaxf(fmaxf(sa[0][j], sa[1][j]), fmaxf(sa[2][j], sa[3][j]));
#pragma unroll
    for (int m = 1; m < 16; m <<= 1)
      cm[j] = fmaxf(cm[j], __shfl_xor(cm[j], m, 32));
  }
  float mnew[8], corr[8], cs[8];
#pragma unroll
  for (int j = 0; j < 8; ++j) {
    mnew[j] = fmaxf(rowm[j], cm[j]);
    corr[j] = __expf(rowm[j] - mnew[j]);
    cs[j] = 0.f;
  }
#pragma unroll
  for (int nt = 0; nt < 4; ++nt)
#pragma unroll
    for (int j = 0; j < 8; ++j) {
      float pv = __expf(sa[nt][j] - mnew[j]);
      sa[nt][j] = pv;
      cs[j] += pv;
    }
#pragma unroll
  for (int j = 0; j < 8; ++j) {
#pragma unroll
    for (int m = 1; m < 16; m <<= 1)
      cs[j] += __shfl_xor(cs[j], m, 32);
    rowl[j] = rowl[j] * corr[j] + cs[j];
    rowm[j] = mnew[j];
  }
#pragma unroll
  for (int nt = 0; nt < 4; ++nt)
#pragma unroll
    for (int j = 0; j < 8; ++j) O[nt][j] *= corr[j];

  // P -> LDS (wave-private rows; DS ops in-order within a wave)
#pragma unroll
  for (int nt = 0; nt < 4; ++nt)
#pragma unroll
    for (int j = 0; j < 8; ++j)
      lP[(w * 16 + j + hl * 8) * 64 + nt * 16 + ln] = f2bf(sa[nt][j]);

#pragma unroll
  for (int kt = 0; kt < 2; ++kt) {
    v16bf pa = ld_fragA(&lP[(w * 16 + ln) * 64 + kt * 32 + hl * 16]);
#pragma unroll
    for (int nt = 0; nt < 4; ++nt) {
      v16bf vb = ld_fragB(&lVt[(nt * 16 + ln) * 64 + kt * 32 + hl * 16]);
      O[nt] = __builtin_amdgcn_wmma_f32_16x16x32_bf16(
          false, pa, false, vb, (short)0, O[nt], false, false);
    }
  }
}

// ---------------------------------------------------------------------------
// Flash attention: one block per (query tile of 64, b, h).  4 waves.
// Async path: double-buffered K/V^T chunks streamed by the async global->LDS
// engine, throttled with s_wait_asynccnt.  Fallback: register-staged pipeline.
// ---------------------------------------------------------------------------
__global__ __launch_bounds__(128) void flash_attn_kernel(const unsigned short* __restrict__ xq,
                                                         const unsigned short* __restrict__ xk,
                                                         const unsigned short* __restrict__ vt,
                                                         unsigned short* __restrict__ ctx) {
  __shared__ __attribute__((aligned(16))) unsigned short lQ[64 * 64];
  __shared__ __attribute__((aligned(16))) unsigned short lP[64 * 64];

  const int tid = threadIdx.x, lane = tid & 31, w = tid >> 5;
  const int ln = lane & 15, hl = lane >> 4;
  const int qt = blockIdx.x;
  const int bh = blockIdx.y;
  const int b  = bh >> 4;
  const int h  = bh & 15;

  // staging geometry: thread covers (row sr, 32 elems at so) of a 64x64 tile
  const int sr = tid >> 1, so = (tid & 1) * 32;
  const unsigned short* gK0 = xk + ((size_t)sr * BS + b) * D_MODEL + h * 64 + so;
  const size_t   gKstep = (size_t)64 * BS * D_MODEL;           // per key chunk
  const unsigned short* gV0 = vt + ((size_t)bh * 64 + sr) * SEQLEN + so;
  const unsigned short* gQ  = xq + ((size_t)(qt * 64 + sr) * BS + b) * D_MODEL + h * 64 + so;

  const v8f zf = {0.f, 0.f, 0.f, 0.f, 0.f, 0.f, 0.f, 0.f};
  v8f O[4] = {zf, zf, zf, zf};
  float rowm[8], rowl[8];
#pragma unroll
  for (int j = 0; j < 8; ++j) { rowm[j] = -1e30f; rowl[j] = 0.f; }
  v16bf qa[2];

#if HAVE_ASYNC_LDS
  __shared__ __attribute__((aligned(16))) unsigned short lK [2][64 * 64];
  __shared__ __attribute__((aligned(16))) unsigned short lVt[2][64 * 64];

  // Q (4 asyncs) then chunk 0 (8 asyncs)
  async_cp32e(gQ, &lQ[sr * 64 + so]);
  async_cp32e(gK0, &lK[0][sr * 64 + so]);
  async_cp32e(gV0, &lVt[0][sr * 64 + so]);
  __builtin_amdgcn_s_wait_asynccnt(8);     // Q resident (chunk0 may still fly)
  __syncthreads();
#pragma unroll
  for (int kt = 0; kt < 2; ++kt)
    qa[kt] = ld_fragA(&lQ[(w * 16 + ln) * 64 + kt * 32 + hl * 16]);

  for (int lc = 0; lc < NCHUNK; ++lc) {
    if (lc + 1 < NCHUNK) {                 // prefetch next chunk, other buffer
      int nb = (lc + 1) & 1;
      async_cp32e(gK0 + (size_t)(lc + 1) * gKstep, &lK[nb][sr * 64 + so]);
      async_cp32e(gV0 + (size_t)(lc + 1) * 64,     &lVt[nb][sr * 64 + so]);
      __builtin_amdgcn_s_wait_asynccnt(8); // chunk lc fully resident
    } else {
      __builtin_amdgcn_s_wait_asynccnt(0);
    }
    __syncthreads();
    attn_chunk(lK[lc & 1], lVt[lc & 1], lP, qa, O, rowm, rowl, w, ln, hl);
    __syncthreads();                       // buffer reusable next iteration
  }
#else
  __shared__ __attribute__((aligned(16))) unsigned short lK [64 * 64];
  __shared__ __attribute__((aligned(16))) unsigned short lVt[64 * 64];

  {  // Q tile, synchronous
    const uint4* src = (const uint4*)gQ;
    uint4* dst = (uint4*)&lQ[sr * 64 + so];
#pragma unroll
    for (int i = 0; i < 4; ++i) dst[i] = src[i];
  }
  // register-staged pipeline for K/V^T
  uint4 rk[4], rv[4];
  {
    const uint4* ks = (const uint4*)gK0;
    const uint4* vs = (const uint4*)gV0;
#pragma unroll
    for (int i = 0; i < 4; ++i) { rk[i] = ks[i]; rv[i] = vs[i]; }
  }
  __syncthreads();
#pragma unroll
  for (int kt = 0; kt < 2; ++kt)
    qa[kt] = ld_fragA(&lQ[(w * 16 + ln) * 64 + kt * 32 + hl * 16]);

  for (int lc = 0; lc < NCHUNK; ++lc) {
    __syncthreads();                       // previous compute done with LDS
    {
      uint4* kd = (uint4*)&lK[sr * 64 + so];
      uint4* vd = (uint4*)&lVt[sr * 64 + so];
#pragma unroll
      for (int i = 0; i < 4; ++i) { kd[i] = rk[i]; vd[i] = rv[i]; }
    }
    __syncthreads();
    if (lc + 1 < NCHUNK) {
      const uint4* ks = (const uint4*)(gK0 + (size_t)(lc + 1) * gKstep);
      const uint4* vs = (const uint4*)(gV0 + (size_t)(lc + 1) * 64);
#pragma unroll
      for (int i = 0; i < 4; ++i) { rk[i] = ks[i]; rv[i] = vs[i]; }
    }
    attn_chunk(lK, lVt, lP, qa, O, rowm, rowl, w, ln, hl);
  }
#endif

#pragma unroll
  for (int j = 0; j < 8; ++j) rowl[j] = 1.f / rowl[j];
#pragma unroll
  for (int nt = 0; nt < 4; ++nt)
#pragma unroll
    for (int j = 0; j < 8; ++j) {
      int srow = qt * 64 + w * 16 + j + hl * 8;
      size_t g = ((size_t)srow * BS + b) * D_MODEL + h * 64 + nt * 16 + ln;
      ctx[g] = f2bf(O[nt][j] * rowl[j]);
    }
}

// ---------------------------------------------------------------------------
extern "C" void kernel_launch(void* const* d_in, const int* in_sizes, int n_in,
                              void* d_out, int out_size, void* d_ws, size_t ws_size,
                              hipStream_t stream) {
  (void)in_sizes; (void)n_in; (void)out_size; (void)ws_size;
  const float* q  = (const float*)d_in[0];
  const float* k  = (const float*)d_in[1];
  const float* v  = (const float*)d_in[2];
  const float* fc = (const float*)d_in[3];
  const float* wq = (const float*)d_in[4];
  const float* wk = (const float*)d_in[5];
  const float* wv = (const float*)d_in[6];
  const float* wo = (const float*)d_in[7];

  const size_t NE = (size_t)MROWS * D_MODEL;
  unsigned short* xq  = (unsigned short*)d_ws;
  unsigned short* xk  = xq + NE;
  unsigned short* xv  = xk + NE;
  unsigned short* ctx = xv + NE;
  unsigned short* vt  = ctx + NE;

  dim3 ggrid(D_MODEL / 128, MROWS / 64);   // (8, 128)
  gemm_nt_wmma<false, false><<<ggrid, 256, 0, stream>>>(q, wq, xq, MROWS, D_MODEL, D_MODEL);
  gemm_nt_wmma<false, false><<<ggrid, 256, 0, stream>>>(k, wk, xk, MROWS, D_MODEL, D_MODEL);
  gemm_nt_wmma<false, false><<<ggrid, 256, 0, stream>>>(v, wv, xv, MROWS, D_MODEL, D_MODEL);

  const int pairs = MROWS * (D_MODEL / 2);
  rope_kernel<<<(pairs + 255) / 256, 256, 0, stream>>>(xq, xk, fc);

  transpose_v_kernel<<<dim3(NCHUNK, BS * NHEADS), 128, 0, stream>>>(xv, vt);

  flash_attn_kernel<<<dim3(SEQLEN / 64, BS * NHEADS), 128, 0, stream>>>(xq, xk, vt, ctx);

  gemm_nt_wmma<true, true><<<ggrid, 256, 0, stream>>>(ctx, wo, d_out, MROWS, D_MODEL, D_MODEL);
}
